// EpisodicMemory_62139586839218
// MI455X (gfx1250) — matrix operand order repaired and checked by reading, not compiled, not run on hardware
//
#include <hip/hip_runtime.h>
#include <hip/hip_bf16.h>

// ---- sizes ----
#define A_N   3
#define T_N   64
#define B_N   256
#define ENTO  2048
#define EM    1024
#define KTOT  5120          // ENTO + 3*EM (unified K space: [x | v0 | v1 | v2])
#define BM    64
#define BN    64
#define KC    32
#define KCP   40            // padded LDS row (80 bytes): conflict-free b128 reads

typedef __attribute__((ext_vector_type(16))) __bf16      v16bf;
typedef __attribute__((ext_vector_type(8)))  float       v8f;
typedef __attribute__((ext_vector_type(4)))  unsigned    u32x4;

union FragU { v16bf v; u32x4 u[2]; };

static __device__ __forceinline__ unsigned short f2bf(float f) {
    unsigned u = __float_as_uint(f);
    u += 0x7FFFu + ((u >> 16) & 1u);          // round-to-nearest-even
    return (unsigned short)(u >> 16);
}

// ---- pack weights fp32 -> bf16 unified-K layout, once ----
__global__ void pack_w(const float* __restrict__ w_ih,   // [3][1024][4096]
                       const float* __restrict__ w_hh,   // [3][1024][1024]
                       unsigned short* __restrict__ Wb)  // [3][1024][5120]
{
    size_t idx = (size_t)blockIdx.x * blockDim.x + threadIdx.x;
    if (idx >= (size_t)A_N * EM * KTOT) return;
    int a   = (int)(idx / ((size_t)EM * KTOT));
    int rem = (int)(idx % ((size_t)EM * KTOT));
    int e = rem / KTOT, k = rem % KTOT;
    float f;
    if (k < ENTO) {
        f = w_ih[((size_t)a * EM + e) * (ENTO + 2 * EM) + k];
    } else {
        int j  = (k - ENTO) >> 10;
        int kk = (k - ENTO) & 1023;
        if (j == a) {
            f = w_hh[((size_t)a * EM + e) * EM + kk];
        } else {
            int pos = (j < a) ? j : j - 1;   // order of "other" attributes
            f = w_ih[((size_t)a * EM + e) * (ENTO + 2 * EM) + ENTO + pos * EM + kk];
        }
    }
    Wb[idx] = f2bf(f);
}

// ---- pack activations [x_t | v0 | v1 | v2] -> bf16, per step ----
__global__ void pack_x(const float* __restrict__ x_t,    // [256][2048]
                       const float* __restrict__ vprev,  // [3][256][1024]
                       unsigned short* __restrict__ Xu)  // [256][5120]
{
    int idx = blockIdx.x * blockDim.x + threadIdx.x;     // < 256*5120
    int b = idx / KTOT, k = idx % KTOT;
    float f;
    if (k < ENTO) {
        f = x_t[(size_t)b * ENTO + k];
    } else {
        int j  = (k - ENTO) >> 10;
        int kk = (k - ENTO) & 1023;
        f = vprev[((size_t)j * B_N + b) * EM + kk];
    }
    Xu[idx] = f2bf(f);
}

// ---- one recurrence step: Y[a] = clamp( Xu @ Wb[a]^T + clamp(vprev[a]) ) ----
__global__ __launch_bounds__(128) void gemm_step(
    const unsigned short* __restrict__ Xu,     // [256][5120] bf16
    const unsigned short* __restrict__ Wb,     // [3][1024][5120] bf16
    const float* __restrict__ vprev,           // [3][256][1024]
    float* __restrict__ out)                   // [3][256][1024]
{
    __shared__ unsigned short As[2][BM][KCP];  // 10 KB
    __shared__ unsigned short Bs[2][BN][KCP];  // 10 KB

    const int a    = blockIdx.z;
    const int m0   = blockIdx.y * BM;
    const int n0   = blockIdx.x * BN;
    const int tid  = threadIdx.x;
    const int wave = tid >> 5;                 // 0..3, grid 2(M)x2(N)
    const int lane = tid & 31;
    const int wM   = (wave >> 1) * 32;
    const int wN   = (wave & 1) * 32;
    const int lr   = lane & 15;
    const int h    = lane >> 4;

    const unsigned short* Wa = Wb + (size_t)a * EM * KTOT;

    // each thread stages 2 uint4 for A and 2 for B per K-chunk
    const int i0 = tid, i1 = tid + 128;        // chunk ids over 256 uint4
    const int ar0 = i0 >> 2, ac0 = (i0 & 3) * 8;
    const int ar1 = i1 >> 2, ac1 = (i1 & 3) * 8;

    v8f acc[2][2];
    #pragma unroll
    for (int i = 0; i < 2; ++i)
        #pragma unroll
        for (int j = 0; j < 2; ++j)
            acc[i][j] = (v8f)0.0f;

    // prologue: chunk 0 -> LDS buffer 0
    {
        u32x4 ra0 = *(const u32x4*)(Xu + (size_t)(m0 + ar0) * KTOT + ac0);
        u32x4 ra1 = *(const u32x4*)(Xu + (size_t)(m0 + ar1) * KTOT + ac1);
        u32x4 rb0 = *(const u32x4*)(Wa + (size_t)(n0 + ar0) * KTOT + ac0);
        u32x4 rb1 = *(const u32x4*)(Wa + (size_t)(n0 + ar1) * KTOT + ac1);
        *(u32x4*)(&As[0][ar0][ac0]) = ra0;
        *(u32x4*)(&As[0][ar1][ac1]) = ra1;
        *(u32x4*)(&Bs[0][ar0][ac0]) = rb0;
        *(u32x4*)(&Bs[0][ar1][ac1]) = rb1;
    }
    __syncthreads();

    const int nIter = KTOT / KC;               // 160
    for (int kk = 0; kk < nIter; ++kk) {
        const int cur = kk & 1;
        u32x4 ra0, ra1, rb0, rb1;
        const bool more = (kk + 1) < nIter;
        if (more) {
            const int k0 = (kk + 1) * KC;
            ra0 = *(const u32x4*)(Xu + (size_t)(m0 + ar0) * KTOT + k0 + ac0);
            ra1 = *(const u32x4*)(Xu + (size_t)(m0 + ar1) * KTOT + k0 + ac1);
            rb0 = *(const u32x4*)(Wa + (size_t)(n0 + ar0) * KTOT + k0 + ac0);
            rb1 = *(const u32x4*)(Wa + (size_t)(n0 + ar1) * KTOT + k0 + ac1);
        }

        // fragment loads (ISA bf16 A/B VGPR layouts) + 4 WMMAs
        FragU fa[2], fb[2];
        #pragma unroll
        for (int tm = 0; tm < 2; ++tm) {
            const unsigned short* base = &As[cur][wM + tm * 16 + lr][0];
            fa[tm].u[0] = *(const u32x4*)(base + 8 * h);        // K = 8h..8h+7
            fa[tm].u[1] = *(const u32x4*)(base + 16 + 8 * h);   // K = 16+8h..
        }
        #pragma unroll
        for (int tn = 0; tn < 2; ++tn) {
            const unsigned short* base = &Bs[cur][wN + tn * 16 + lr][0];
            fb[tn].u[0] = *(const u32x4*)(base + 8 * h);        // K = 8h..8h+7
            fb[tn].u[1] = *(const u32x4*)(base + 16 + 8 * h);   // K = 16+8h..
        }
        #pragma unroll
        for (int tm = 0; tm < 2; ++tm)
            #pragma unroll
            for (int tn = 0; tn < 2; ++tn)
                acc[tm][tn] = __builtin_amdgcn_wmma_f32_16x16x32_bf16(
                    false, fa[tm].v, false, fb[tn].v,
                    (short)0, acc[tm][tn], false, false);

        if (more) {
            const int nxt = cur ^ 1;
            *(u32x4*)(&As[nxt][ar0][ac0]) = ra0;
            *(u32x4*)(&As[nxt][ar1][ac1]) = ra1;
            *(u32x4*)(&Bs[nxt][ar0][ac0]) = rb0;
            *(u32x4*)(&Bs[nxt][ar1][ac1]) = rb1;
            __syncthreads();
        }
    }

    // epilogue: v_next = clamp(acc + clamp(v_prev))
    #pragma unroll
    for (int tm = 0; tm < 2; ++tm)
        #pragma unroll
        for (int tn = 0; tn < 2; ++tn)
            #pragma unroll
            for (int r = 0; r < 8; ++r) {
                int mm = m0 + wM + tm * 16 + r + 8 * h;
                int nn = n0 + wN + tn * 16 + lr;
                size_t o = ((size_t)a * B_N + mm) * EM + nn;
                float vp = vprev[o];
                vp = fminf(fmaxf(vp, -1.0f), 1.0f);
                float y = acc[tm][tn][r] + vp;
                out[o] = fminf(fmaxf(y, -1.0f), 1.0f);
            }
}

extern "C" void kernel_launch(void* const* d_in, const int* in_sizes, int n_in,
                              void* d_out, int out_size, void* d_ws, size_t ws_size,
                              hipStream_t stream) {
    const float* ento = (const float*)d_in[0];   // [64][256][2048]
    const float* v0   = (const float*)d_in[1];   // [3][256][1024]
    const float* wih  = (const float*)d_in[2];   // [3][1024][4096]
    const float* whh  = (const float*)d_in[3];   // [3][1024][1024]
    float* out = (float*)d_out;                  // [64][3][256][1024]

    unsigned short* Wb = (unsigned short*)d_ws;                 // 31.5 MB bf16
    unsigned short* Xu = Wb + (size_t)A_N * EM * KTOT;          // +2.6 MB bf16

    const size_t wtot = (size_t)A_N * EM * KTOT;
    pack_w<<<(unsigned)((wtot + 255) / 256), 256, 0, stream>>>(wih, whh, Wb);

    const size_t stepElems = (size_t)A_N * B_N * EM;            // 786432
    const float* vprev = v0;
    for (int t = 0; t < T_N; ++t) {
        pack_x<<<(B_N * KTOT) / 256, 256, 0, stream>>>(
            ento + (size_t)t * B_N * ENTO, vprev, Xu);
        dim3 grid(EM / BN, B_N / BM, A_N);                      // 16 x 4 x 3
        gemm_step<<<grid, 128, 0, stream>>>(Xu, Wb, vprev,
                                            out + (size_t)t * stepElems);
        vprev = out + (size_t)t * stepElems;
    }
}
